// SpikingLinearLayer_2843268350005
// MI455X (gfx1250) — matrix operand order repaired and checked by reading, not compiled, not run on hardware
//
#include <hip/hip_runtime.h>

// ---------------------------------------------------------------------------
// Spiking linear layer (LIF), fused WMMA GEMM + recurrence for gfx1250.
//   x:      [T=20, B=2048, DIN=1024] fp32 binary spikes   (read ONCE)
//   weight: [DOUT=512, DIN=1024]     fp32  -> f16 in ws   (lives in L2)
//   out:    [T, B, DOUT]             fp32 spikes          (NT stores)
// ---------------------------------------------------------------------------

typedef __attribute__((ext_vector_type(16))) _Float16 v16h;
typedef __attribute__((ext_vector_type(8)))  _Float16 v8h;
typedef __attribute__((ext_vector_type(4)))  _Float16 v4h;
typedef __attribute__((ext_vector_type(8)))  float    v8f;

constexpr int   T    = 20;
constexpr int   Bsz  = 2048;
constexpr int   DIN  = 1024;
constexpr int   DOUT = 512;
constexpr float DEC_A = 1.0f / 20.0f;  // dt/tau_m
constexpr float DEC_B = 1.0f / 5.0f;   // dt/tau_s

constexpr int BM      = 16;           // batch rows per workgroup (one M tile)
constexpr int WAVES   = 8;            // wave32s per workgroup
constexpr int NPW     = 64;           // output columns per wave
constexpr int NT      = NPW / 16;     // 4 N-tiles per wave -> 4 indep. WMMA chains
constexpr int LDS_STR = DIN + 8;      // padded f16 row stride (bank-conflict dodge)

// --- Phase 0: weight fp32 -> f16 (stays hot in L2 afterwards) --------------
__global__ __launch_bounds__(256) void weight_to_f16(const float* __restrict__ w,
                                                     _Float16* __restrict__ wh) {
  int i = (blockIdx.x * blockDim.x + threadIdx.x) * 4;
  float4 v = *(const float4*)(w + i);
  v4h h = { (_Float16)v.x, (_Float16)v.y, (_Float16)v.z, (_Float16)v.w };
  *(v4h*)(wh + i) = h;
}

// --- Phase 1: fused GEMM + LIF scan ----------------------------------------
__global__ __launch_bounds__(256) void snn_lif(const float* __restrict__ x,
                                               const _Float16* __restrict__ wh,
                                               float* __restrict__ out) {
  __shared__ _Float16 sx[BM][LDS_STR];   // 16 x 1032 f16 = 33 KB

  const int tid   = threadIdx.x;
  const int wave  = tid >> 5;
  const int lane  = tid & 31;
  const int lo    = lane & 15;           // N (B frag) / M (A frag) within tile
  const int hi8   = (lane >> 4) << 3;    // C layout: lanes 16-31 -> rows M+8
  const int brow0 = blockIdx.x * BM;
  const int nbase = wave * NPW;          // one WG covers all 512 output cols

  // Membrane potential / synaptic current, registers across all timesteps.
  v8f V[NT], I[NT];
#pragma unroll
  for (int nt = 0; nt < NT; ++nt) { V[nt] = (v8f)0.0f; I[nt] = (v8f)0.0f; }

  for (int t = 0; t < T; ++t) {
    // ---- stage x[t, brow0:brow0+16, :] into LDS as f16 (coalesced) ----
    const size_t xbase = ((size_t)t * Bsz + brow0) * DIN;
#pragma unroll
    for (int j = 0; j < (BM * DIN) / (256 * 4); ++j) {   // 16 iterations
      int flat = (j * 256 + tid) * 4;
      int row  = flat >> 10;            // / DIN
      int col  = flat & (DIN - 1);
      float4 v = *(const float4*)(x + xbase + flat);
      v4h h = { (_Float16)v.x, (_Float16)v.y, (_Float16)v.z, (_Float16)v.w };
      *(v4h*)&sx[row][col] = h;
    }
    __syncthreads();

    // ---- L2 prefetch of next timestep's x tile (global_prefetch_b8):
    //      overlaps HBM latency with the WMMA loop at zero register cost.
    if (t + 1 < T) {
      const float* nx = x + xbase + (size_t)Bsz * DIN;
#pragma unroll
      for (int j = 0; j < (BM * DIN) / (256 * 4); ++j) {
        __builtin_prefetch(nx + (j * 256 + tid) * 4, 0, 1);
      }
    }

    // ---- Z_t tile = x_tile @ W^T via v_wmma_f32_16x16x32_f16 ----
    v8f acc[NT];
#pragma unroll
    for (int nt = 0; nt < NT; ++nt) acc[nt] = (v8f)0.0f;

#pragma unroll 2
    for (int kt = 0; kt < DIN / 32; ++kt) {
      // A fragment 16x32 f16: lanes 0-15 hold K[0..7]+[16..23], lanes 16-31
      // hold K[8..15]+[24..31] of row M=lane%16 (ISA 7.12.2 layout).
      const int klo = kt * 32 + ((lane < 16) ? 0 : 8);
      v8h a0 = *(const v8h*)&sx[lo][klo];
      v8h a1 = *(const v8h*)&sx[lo][klo + 16];
      v16h a;
#pragma unroll
      for (int i = 0; i < 8; ++i) { a[i] = a0[i]; a[i + 8] = a1[i]; }

#pragma unroll
      for (int nt = 0; nt < NT; ++nt) {
        // B fragment 32x16 f16: lane holds column N=lane%16, K in the same
        // order as A (low lanes K[0..7]+[16..23], high lanes K[8..15]+[24..31])
        // from row-major W[n, k], so two global_load_b128 from L2-resident weights.
        const _Float16* wp = wh + (size_t)(nbase + nt * 16 + lo) * DIN
                                + kt * 32 + ((lane < 16) ? 0 : 8);
        v8h b0 = *(const v8h*)wp;
        v8h b1 = *(const v8h*)(wp + 16);
        v16h b;
#pragma unroll
        for (int i = 0; i < 8; ++i) { b[i] = b0[i]; b[i + 8] = b1[i]; }

        acc[nt] = __builtin_amdgcn_wmma_f32_16x16x32_f16(
            false, a, false, b, (short)0, acc[nt], false, false);
      }
    }

    // ---- LIF recurrence + spike emission (C layout: VGPR r -> M=r+hi8) ----
#pragma unroll
    for (int nt = 0; nt < NT; ++nt) {
      const int ncol = nbase + nt * 16 + lo;
      float* op = out + ((size_t)t * Bsz + brow0 + hi8) * DOUT + ncol;
#pragma unroll
      for (int r = 0; r < 8; ++r) {
        float v  = V[nt][r];
        float ii = I[nt][r];
        v  = (1.0f - DEC_A) * v + DEC_A * ii;       // integrate membrane (old I)
        ii = (1.0f - DEC_B) * ii + acc[nt][r];      // integrate synaptic current
        float s = (v > 1.0f) ? 1.0f : 0.0f;         // threshold
        V[nt][r] = (v > 1.0f) ? 0.0f : v;           // reset fired neurons
        I[nt][r] = ii;
        // write-once spikes: non-temporal so they don't evict W from L2
        __builtin_nontemporal_store(s, op + (size_t)r * DOUT);
      }
    }
    __syncthreads();   // protect LDS before next timestep's staging
  }
}

extern "C" void kernel_launch(void* const* d_in, const int* in_sizes, int n_in,
                              void* d_out, int out_size, void* d_ws, size_t ws_size,
                              hipStream_t stream) {
  const float* x = (const float*)d_in[0];        // [T, B, DIN] fp32
  const float* w = (const float*)d_in[1];        // [DOUT, DIN] fp32
  float* out     = (float*)d_out;                // [T, B, DOUT] fp32
  _Float16* wh   = (_Float16*)d_ws;              // 1 MB f16 weights

  (void)in_sizes; (void)n_in; (void)out_size; (void)ws_size;

  // W fp32 -> f16 (524288 elems, 4 per thread)
  weight_to_f16<<<(DOUT * DIN) / (256 * 4), 256, 0, stream>>>(w, wh);

  // Fused SNN: 128 workgroups x 8 waves; each WG owns 16 batch rows x all 512 cols
  snn_lif<<<dim3(Bsz / BM), WAVES * 32, 0, stream>>>(x, wh, out);
}
